// IndependentCapsuleNetwork_11982958756231
// MI455X (gfx1250) — matrix operand order repaired and checked by reading, not compiled, not run on hardware
//
#include <hip/hip_runtime.h>

// ---------------------------------------------------------------------------
// CapsNet forward for gfx1250 (MI455X): bf16 WMMA implicit-GEMM convs with
// async-to-LDS weight staging + double-buffered software pipeline, and
// bandwidth-minimal routing. All kernels launched on `stream`; scratch in d_ws.
// ---------------------------------------------------------------------------

typedef __bf16 bf16;
typedef __attribute__((ext_vector_type(16))) __bf16        v16bf;
typedef __attribute__((ext_vector_type(8)))  float         v8f;
typedef __attribute__((ext_vector_type(4)))  unsigned int  u32x4;
typedef __attribute__((ext_vector_type(4)))  int           i32x4;
typedef __attribute__((ext_vector_type(4)))  float         f32x4;

#define LDSA 40   // 32 + 8 pad (bf16) -> conflict-free ds_load_b128
#define LDSB 40

union BFPack8 { u32x4 q; bf16 h[8]; };

// ---------------- CDNA5 async global->LDS helpers --------------------------
// GLOBAL_LOAD_ASYNC_TO_LDS_B128 (ASYNCcnt, ISA 15.18.3 op 98). Builtin
// signature (from hipcc diagnostic): (int4 global*, int4 local*, imm, imm).

#if __has_builtin(__builtin_amdgcn_global_load_async_to_lds_b128)
#define USE_ASYNC_BUILTIN 1
#else
#define USE_ASYNC_BUILTIN 0
#endif
#if __has_builtin(__builtin_amdgcn_s_wait_asynccnt)
#define USE_WAITA_BUILTIN 1
#else
#define USE_WAITA_BUILTIN 0
#endif

__device__ __forceinline__ void async_copy16(const bf16* gsrc, bf16* ldst) {
#if USE_ASYNC_BUILTIN
  __builtin_amdgcn_global_load_async_to_lds_b128(
      (__attribute__((address_space(1))) i32x4*)gsrc,
      (__attribute__((address_space(3))) i32x4*)ldst, 0, 0);
#else
  unsigned loff = (unsigned)(unsigned long long)
      (__attribute__((address_space(3))) void*)ldst;
  unsigned long long ga = (unsigned long long)gsrc;
  asm volatile("global_load_async_to_lds_b128 %0, %1, off"
               :: "v"(loff), "v"(ga) : "memory");
#endif
}

__device__ __forceinline__ void wait_async0() {
#if USE_WAITA_BUILTIN
  __builtin_amdgcn_s_wait_asynccnt(0);
#else
  asm volatile("s_wait_asynccnt 0x0" ::: "memory");
#endif
}

// ------------------------- small prep kernels ------------------------------

// repack conv weights fp32 [OC][Ktot] -> bf16 [OC][Kpad], zero tail.
// Kpad % 32 == 0 so every K-step tile is fully valid and rows stay 16B aligned.
__global__ void repack_w_kernel(const float* __restrict__ src, bf16* __restrict__ dst,
                                int OC, int Ktot, int Kpad) {
  int i = blockIdx.x * blockDim.x + threadIdx.x;
  if (i >= OC * Kpad) return;
  int oc = i / Kpad, k = i - oc * Kpad;
  float v = (k < Ktot) ? src[(size_t)oc * Ktot + k] : 0.f;
  dst[i] = (bf16)v;
}

__global__ void bnfold_kernel(const float* __restrict__ g, const float* __restrict__ b,
                              const float* __restrict__ m, const float* __restrict__ v,
                              const float* __restrict__ bias,
                              float* __restrict__ scale, float* __restrict__ shift, int C) {
  int c = blockIdx.x * blockDim.x + threadIdx.x;
  if (c < C) {
    float s = g[c] * rsqrtf(v[c] + 1e-5f);
    scale[c] = s;
    shift[c] = (bias[c] - m[c]) * s + b[c];
  }
}

__global__ void pcfold_kernel(const float* __restrict__ bias,
                              float* __restrict__ scale, float* __restrict__ shift, int C) {
  int c = blockIdx.x * blockDim.x + threadIdx.x;
  if (c < C) { scale[c] = 1.0f; shift[c] = bias[c]; }
}

__global__ void gray_kernel(const float* __restrict__ x, bf16* __restrict__ g,
                            int B, int HW) {
  int i = blockIdx.x * blockDim.x + threadIdx.x;
  if (i >= B * HW) return;
  int b = i / HW, p = i - b * HW;
  const float* xb = x + (size_t)b * 3 * HW;
  g[i] = (bf16)(0.299f * xb[p] + 0.587f * xb[HW + p] + 0.114f * xb[2 * HW + p]);
}

// ------------------- implicit-GEMM conv via WMMA bf16 ----------------------
// M = OC, N = B*OH*OW, K = IC*81 (padded to 32). Block: 256 thr = 8 waves,
// tile 64M x 128N, K-step 32. Wave (wm 0..3, wn 0..1) owns 16M x 64N as four
// 16x16 f32 accumulators (one A fragment reused across 4 WMMAs). A staged via
// async global->LDS b128 DMA; B (im2col gather) pipelined through registers.
// Double-buffered LDS; one barrier per K-step.

template<int CSTRIDE, bool RELU, bool STORE_F32>
__global__ __launch_bounds__(256) void conv_wmma_kernel(
    const bf16* __restrict__ in, const bf16* __restrict__ wgt,   // wgt: [OC][Kpad]
    const float* __restrict__ scale, const float* __restrict__ shift,
    void* __restrict__ outp,
    int B, int IC, int OC, int IH, int IW, int OH, int OW) {
  const int Ktot = IC * 81;
  const int Kpad = (Ktot + 31) & ~31;
  const int OHW  = OH * OW;
  const int Ntot = B * OHW;

  __shared__ __align__(16) bf16 Abuf[2][64 * LDSA];
  __shared__ __align__(16) bf16 Bbuf[2][128 * LDSB];

  const int t    = threadIdx.x;
  const int lane = t & 31, wave = t >> 5;
  const int wm   = wave >> 1;            // 0..3 : 16-row band
  const int wn   = wave & 1;             // 0..1 : 64-col band
  const int grp  = lane >> 4, lid = lane & 15;
  const int ocbase = blockIdx.y * 64;
  const int nbase  = blockIdx.x * 128;

  v8f acc0 = {0.f,0.f,0.f,0.f,0.f,0.f,0.f,0.f};
  v8f acc1 = acc0, acc2 = acc0, acc3 = acc0;

  // ---- A staging assignment: thread -> row t/4, 16B chunk (t%4)*8 ----
  const int arow = t >> 2;
  const int achk = (t & 3) << 3;
  const bf16* agsrc = wgt + (size_t)(ocbase + arow) * Kpad + achk;
  bf16* aldst[2] = { &Abuf[0][arow * LDSA + achk], &Abuf[1][arow * LDSA + achk] };

  // ---- B staging assignment: thread -> n_local t/2, 16 k at (t%2)*16 ----
  const int bnl = t >> 1;
  const int bkc = (t & 1) << 4;
  const int ng  = nbase + bnl;
  int ib = 0, ih0 = 0, iw0 = 0;
  const bool nv = (ng < Ntot);
  if (nv) {
    ib = ng / OHW;
    int r  = ng - ib * OHW;
    int oh = r / OW;
    ih0 = oh * CSTRIDE;
    iw0 = (r - oh * OW) * CSTRIDE;
  }

  const int ksteps = Kpad >> 5;
  BFPack8 breg[2];

  // im2col gather of 16 consecutive k for this thread into registers
  auto load_breg = [&](int kb) {
#pragma unroll
    for (int j = 0; j < 16; ++j) {
      int k = kb + bkc + j;
      bf16 v = (bf16)0.f;
      if (nv && k < Ktot) {
        int ic = k / 81; int r = k - ic * 81;
        int kh = r / 9;  int kw = r - kh * 9;
        v = in[(((size_t)ib * IC + ic) * IH + (ih0 + kh)) * IW + (iw0 + kw)];
      }
      breg[j >> 3].h[j & 7] = v;
    }
  };
  auto store_breg = [&](int buf) {
    bf16* bd = &Bbuf[buf][bnl * LDSB + bkc];
    *reinterpret_cast<u32x4*>(bd)     = breg[0].q;
    *reinterpret_cast<u32x4*>(bd + 8) = breg[1].q;
  };

  // ---- prologue: stage tile 0 ----
  async_copy16(agsrc, aldst[0]);
  load_breg(0);
  store_breg(0);
  wait_async0();
  __syncthreads();

  for (int ks = 0; ks < ksteps; ++ks) {
    const int cur = ks & 1, nxt = cur ^ 1;
    const bool more = (ks + 1) < ksteps;
    if (more) {
      async_copy16(agsrc + ((ks + 1) << 5), aldst[nxt]);
      load_breg((ks + 1) << 5);
    }

    // fragments per ISA 16-bit WMMA lane layouts; A reused across 4 WMMAs
    v16bf af;
    {
      u32x4* ap = reinterpret_cast<u32x4*>(&af);
      const bf16* as = &Abuf[cur][(wm * 16 + lid) * LDSA + (grp << 3)];
      ap[0] = *reinterpret_cast<const u32x4*>(as);        // K = 8g .. 8g+7
      ap[1] = *reinterpret_cast<const u32x4*>(as + 16);   // K = 8g+16 .. 8g+23
    }
#pragma unroll
    for (int s = 0; s < 4; ++s) {
      v16bf bfm;
      u32x4* bp = reinterpret_cast<u32x4*>(&bfm);
      const bf16* bs = &Bbuf[cur][(wn * 64 + s * 16 + lid) * LDSB + (grp << 3)];
      bp[0] = *reinterpret_cast<const u32x4*>(bs);        // K = 8g .. 8g+7
      bp[1] = *reinterpret_cast<const u32x4*>(bs + 16);   // K = 8g+16 .. 8g+23
      v8f& a = (s == 0) ? acc0 : (s == 1) ? acc1 : (s == 2) ? acc2 : acc3;
      a = __builtin_amdgcn_wmma_f32_16x16x32_bf16(false, af, false, bfm,
                                                  (short)0, a, false, false);
    }

    if (more) {
      store_breg(nxt);
      wait_async0();
    }
    __syncthreads();
  }

  // ---- epilogue: fused BN (+bias), optional ReLU; D: m = r + 8*grp, n = lid
#pragma unroll
  for (int s = 0; s < 4; ++s) {
    const v8f& a = (s == 0) ? acc0 : (s == 1) ? acc1 : (s == 2) ? acc2 : acc3;
    const int ns = nbase + wn * 64 + s * 16 + lid;
    if (ns < Ntot) {
      int ob = ns / OHW; int r = ns - ob * OHW;
      int oh = r / OW, ow = r - oh * OW;
#pragma unroll
      for (int rr = 0; rr < 8; ++rr) {
        int oc = ocbase + wm * 16 + (grp << 3) + rr;
        float v = a[rr] * scale[oc] + shift[oc];
        if (RELU) v = fmaxf(v, 0.f);
        size_t oi = (((size_t)ob * OC + oc) * OH + oh) * OW + ow;
        if (STORE_F32) reinterpret_cast<float*>(outp)[oi] = v;
        else           reinterpret_cast<bf16*>(outp)[oi] = (bf16)v;
      }
    }
  }
}

// ------------------------- capsule-side kernels ----------------------------

__global__ void squash_u_kernel(const float* __restrict__ y, bf16* __restrict__ u,
                                int B, int N, int P) {
  int i = blockIdx.x * blockDim.x + threadIdx.x;
  if (i >= B * N) return;
  int b = i / N, n = i - b * N;
  int g = n / P, p = n - g * P;
  const float* base = y + ((size_t)b * 256 + g * 8) * P + p;
  float vals[8], sn = 0.f;
#pragma unroll
  for (int e = 0; e < 8; ++e) { vals[e] = base[(size_t)e * P]; sn += vals[e] * vals[e]; }
  float sc = sn / (1.f + sn) * rsqrtf(sn + 1e-8f);
#pragma unroll
  for (int e = 0; e < 8; ++e) u[(size_t)i * 8 + e] = (bf16)(vals[e] * sc);
}

__global__ __launch_bounds__(96) void uhat_kernel(const float* __restrict__ W,
                                                  const bf16* __restrict__ u,
                                                  bf16* __restrict__ uh,
                                                  int B, int N) {
  int n = blockIdx.x;
  int t = threadIdx.x;
  __shared__ float ush[96];  // [b][d] = 12*8
  if (t < 96) ush[t] = (float)u[((size_t)(t >> 3) * N + n) * 8 + (t & 7)];
  __syncthreads();
  if (t < 80) {
    const float* wr = W + ((size_t)n * 80 + t) * 8;
    f32x4 w0 = *reinterpret_cast<const f32x4*>(wr);
    f32x4 w1 = *reinterpret_cast<const f32x4*>(wr + 4);
#pragma unroll
    for (int b = 0; b < 12; ++b) {
      float s = 0.f;
#pragma unroll
      for (int d = 0; d < 4; ++d) s += w0[d] * ush[b * 8 + d] + w1[d] * ush[b * 8 + 4 + d];
      uh[((size_t)b * N + n) * 80 + t] = (bf16)s;
    }
  }
}

__global__ __launch_bounds__(256) void route_s_kernel(
    const bf16* __restrict__ uh, const float* __restrict__ vhist,
    float* __restrict__ partial, int iter, int B, int N, int nblk) {
  const int b = blockIdx.y, blk = blockIdx.x;
  const int t = threadIdx.x;
  const int lane = t & 31, wave = t >> 5;

  __shared__ float vsh[2][80];
  __shared__ float wred[8][80];
  if (t < 160) {
    int s = t / 80, i2 = t - s * 80;
    float val = 0.f;
    if (s < iter) val = vhist[((size_t)s * B + b) * 80 + i2];
    vsh[s][i2] = val;
  }
  __syncthreads();

  float acc[80];
#pragma unroll
  for (int i = 0; i < 80; ++i) acc[i] = 0.f;

  for (int rep = 0; rep < 2; ++rep) {            // 512 capsules per block
    int n = blk * 512 + rep * 256 + t;
    if (n >= N) continue;
    const u32x4* up4 = reinterpret_cast<const u32x4*>(uh + ((size_t)b * N + n) * 80);
    float uhv[80];
#pragma unroll
    for (int c = 0; c < 10; ++c) {
      BFPack8 pk; pk.q = up4[c];
#pragma unroll
      for (int j = 0; j < 8; ++j) uhv[c * 8 + j] = (float)pk.h[j];
    }
    float blog[5] = {0.f, 0.f, 0.f, 0.f, 0.f};
    if (iter >= 1) {
#pragma unroll
      for (int o = 0; o < 5; ++o) {
        float a = 0.f;
#pragma unroll
        for (int e = 0; e < 16; ++e) a += uhv[o * 16 + e] * vsh[0][o * 16 + e];
        blog[o] += a;
      }
    }
    if (iter >= 2) {
#pragma unroll
      for (int o = 0; o < 5; ++o) {
        float a = 0.f;
#pragma unroll
        for (int e = 0; e < 16; ++e) a += uhv[o * 16 + e] * vsh[1][o * 16 + e];
        blog[o] += a;
      }
    }
    float mx = blog[0];
#pragma unroll
    for (int o = 1; o < 5; ++o) mx = fmaxf(mx, blog[o]);
    float ex[5], se = 0.f;
#pragma unroll
    for (int o = 0; o < 5; ++o) { ex[o] = __expf(blog[o] - mx); se += ex[o]; }
    float inv = 1.f / se;
#pragma unroll
    for (int o = 0; o < 5; ++o) {
      float c = ex[o] * inv;
#pragma unroll
      for (int e = 0; e < 16; ++e) acc[o * 16 + e] += c * uhv[o * 16 + e];
    }
  }

  // wave32 butterfly reduce, then cross-wave via LDS
#pragma unroll
  for (int i = 0; i < 80; ++i) {
    float v = acc[i];
    v += __shfl_xor(v, 16, 32);
    v += __shfl_xor(v, 8, 32);
    v += __shfl_xor(v, 4, 32);
    v += __shfl_xor(v, 2, 32);
    v += __shfl_xor(v, 1, 32);
    if (lane == 0) wred[wave][i] = v;
  }
  __syncthreads();
  if (t < 80) {
    float s = 0.f;
#pragma unroll
    for (int w = 0; w < 8; ++w) s += wred[w][t];
    partial[((size_t)b * nblk + blk) * 80 + t] = s;
  }
}

__global__ void route_v_kernel(const float* __restrict__ partial,
                               float* __restrict__ vhist,
                               int iter, int B, int nblk) {
  int b = blockIdx.x;
  int t = threadIdx.x;
  bool act = t < 80;
  __shared__ float sq[80];
  float s = 0.f;
  if (act) {
    for (int k = 0; k < nblk; ++k) s += partial[((size_t)b * nblk + k) * 80 + t];
    sq[t] = s * s;
  }
  __syncthreads();
  if (act) {
    int o = t >> 4;
    float sn = 0.f;
#pragma unroll
    for (int e = 0; e < 16; ++e) sn += sq[o * 16 + e];
    float sc = sn / (1.f + sn) * rsqrtf(sn + 1e-8f);
    vhist[((size_t)iter * B + b) * 80 + t] = s * sc;
  }
}

__global__ void finalize_kernel(const float* __restrict__ vhist,
                                float* __restrict__ out, int B) {
  const float* v = vhist + (size_t)2 * B * 80;   // final-iteration v
  int t = threadIdx.x;
  if (t < B * 80) out[B * 5 + t] = v[t];         // v after probs
  if (t < B * 5) {
    float sn = 0.f;
#pragma unroll
    for (int e = 0; e < 16; ++e) { float x = v[t * 16 + e]; sn += x * x; }
    out[t] = sqrtf(sn);
  }
}

// ------------------------------- host --------------------------------------

extern "C" void kernel_launch(void* const* d_in, const int* in_sizes, int n_in,
                              void* d_out, int out_size, void* d_ws, size_t ws_size,
                              hipStream_t stream) {
  (void)in_sizes; (void)n_in; (void)out_size; (void)ws_size;
  const float* x    = (const float*)d_in[0];
  const float* c1w  = (const float*)d_in[1];
  const float* c1b  = (const float*)d_in[2];
  const float* bn1g = (const float*)d_in[3];
  const float* bn1b = (const float*)d_in[4];
  const float* bn1m = (const float*)d_in[5];
  const float* bn1v = (const float*)d_in[6];
  const float* c2w  = (const float*)d_in[7];
  const float* c2b  = (const float*)d_in[8];
  const float* bn2g = (const float*)d_in[9];
  const float* bn2b = (const float*)d_in[10];
  const float* bn2m = (const float*)d_in[11];
  const float* bn2v = (const float*)d_in[12];
  const float* c3w  = (const float*)d_in[13];
  const float* c3b  = (const float*)d_in[14];
  const float* bn3g = (const float*)d_in[15];
  const float* bn3b = (const float*)d_in[16];
  const float* bn3m = (const float*)d_in[17];
  const float* bn3v = (const float*)d_in[18];
  const float* pcw  = (const float*)d_in[19];
  const float* pcb  = (const float*)d_in[20];
  const float* W    = (const float*)d_in[21];
  float* out = (float*)d_out;

  const int Bn = 12, N = 61952, NBLK = 121;

  char* ws = (char*)d_ws;
  size_t off = 0;
  auto take = [&](size_t bytes) -> char* {
    char* p = ws + off;
    off = (off + bytes + 255) & ~(size_t)255;
    return p;
  };

  // K-padded bf16 weights (Kpad = round_up(IC*81, 32))
  bf16* w1 = (bf16*)take((size_t)64 * 96 * 2);         // Ktot 81  -> Kpad 96
  bf16* w2 = (bf16*)take((size_t)128 * 5184 * 2);      // already %32
  bf16* w3 = (bf16*)take((size_t)256 * 10368 * 2);
  bf16* wp = (bf16*)take((size_t)256 * 20736 * 2);
  float* sc1 = (float*)take(64 * 4);  float* sh1 = (float*)take(64 * 4);
  float* sc2 = (float*)take(128 * 4); float* sh2 = (float*)take(128 * 4);
  float* sc3 = (float*)take(256 * 4); float* sh3 = (float*)take(256 * 4);
  float* scp = (float*)take(256 * 4); float* shp = (float*)take(256 * 4);
  // ping-pong activations: act0 = gray -> a2 -> pc(f32); act1 = a1 -> a3
  char* act0 = take((size_t)33214464);  // >= max(gray 1.2MB, a2 33.2MB, pc f32 23.8MB)
  char* act1 = take((size_t)71663616);  // >= max(a1 71.7MB, a3 56.6MB)
  bf16*  u    = (bf16*)take((size_t)Bn * N * 8 * 2);
  bf16*  uhat = (bf16*)take((size_t)Bn * N * 80 * 2);
  float* vhist   = (float*)take((size_t)3 * Bn * 80 * 4);
  float* partial = (float*)take((size_t)Bn * NBLK * 80 * 4);

  // fold BN + bias; repack weights to bf16 with padded K
  bnfold_kernel<<<1, 64, 0, stream>>>(bn1g, bn1b, bn1m, bn1v, c1b, sc1, sh1, 64);
  bnfold_kernel<<<2, 64, 0, stream>>>(bn2g, bn2b, bn2m, bn2v, c2b, sc2, sh2, 128);
  bnfold_kernel<<<4, 64, 0, stream>>>(bn3g, bn3b, bn3m, bn3v, c3b, sc3, sh3, 256);
  pcfold_kernel<<<4, 64, 0, stream>>>(pcb, scp, shp, 256);
  repack_w_kernel<<<(64 * 96 + 255) / 256, 256, 0, stream>>>(c1w, w1, 64, 81, 96);
  repack_w_kernel<<<(128 * 5184 + 255) / 256, 256, 0, stream>>>(c2w, w2, 128, 5184, 5184);
  repack_w_kernel<<<(256 * 10368 + 255) / 256, 256, 0, stream>>>(c3w, w3, 256, 10368, 10368);
  repack_w_kernel<<<(256 * 20736 + 255) / 256, 256, 0, stream>>>(pcw, wp, 256, 20736, 20736);

  // grayscale
  gray_kernel<<<(12 * 224 * 224 + 255) / 256, 256, 0, stream>>>(x, (bf16*)act0, Bn, 224 * 224);

  // conv stack (implicit GEMM, WMMA bf16, 64M x 128N blocks)
  conv_wmma_kernel<1, true, false><<<dim3(4374, 1), 256, 0, stream>>>(
      (const bf16*)act0, w1, sc1, sh1, act1, Bn, 1, 64, 224, 224, 216, 216);
  conv_wmma_kernel<2, true, false><<<dim3(1014, 2), 256, 0, stream>>>(
      (const bf16*)act1, w2, sc2, sh2, act0, Bn, 64, 128, 216, 216, 104, 104);
  conv_wmma_kernel<1, true, false><<<dim3(864, 4), 256, 0, stream>>>(
      (const bf16*)act0, w3, sc3, sh3, act1, Bn, 128, 256, 104, 104, 96, 96);
  conv_wmma_kernel<2, false, true><<<dim3(182, 4), 256, 0, stream>>>(
      (const bf16*)act1, wp, scp, shp, act0, Bn, 256, 256, 96, 96, 44, 44);

  // capsules
  squash_u_kernel<<<(Bn * N + 255) / 256, 256, 0, stream>>>((const float*)act0, u, Bn, N, 1936);
  uhat_kernel<<<N, 96, 0, stream>>>(W, u, uhat, Bn, N);

  // dynamic routing (3 iterations), b_logit recomputed from v history
  for (int it = 0; it < 3; ++it) {
    route_s_kernel<<<dim3(NBLK, Bn), 256, 0, stream>>>(uhat, vhist, partial, it, Bn, N, NBLK);
    route_v_kernel<<<Bn, 96, 0, stream>>>(partial, vhist, it, Bn, NBLK);
  }
  finalize_kernel<<<1, 1024, 0, stream>>>(vhist, out, Bn);
}
